// AttentionHead_32263794328439
// MI455X (gfx1250) — matrix operand order repaired and checked by reading, not compiled, not run on hardware
//
#include <hip/hip_runtime.h>
#include <stdint.h>

// CDNA5 (gfx1250) flash-attention with length masking.
// wave32; V_WMMA_F32_16X16X32_F16 for QK^T, P*V, and P*1 (row sums).
// Fast path: pre-pass converts K/V to f16 (V pre-transposed per 32-key tile) and
// computes per-batch mean(V) (the exact output for fully-masked rows).
// Main loop double-buffers K/V tiles into LDS via GLOBAL_LOAD_ASYNC_TO_LDS_B128,
// iterates only ceil(len/32) key tiles, applies the column mask only on the tail
// tile, and uses DPP ROW_XMASK for the softmax max-reduction (no LDS round trips).

typedef _Float16 v16h __attribute__((ext_vector_type(16)));
typedef float    v8f  __attribute__((ext_vector_type(8)));
typedef _Float16 h2   __attribute__((ext_vector_type(2)));
typedef _Float16 h4   __attribute__((ext_vector_type(4)));
typedef int      v4i  __attribute__((ext_vector_type(4)));

#define HEAD_D 128

// ---- 16-lane butterfly max via DPP16 ROW_XMASK (0x160 | xor_mask) ----
template <int CTRL>
__device__ __forceinline__ float dpp_max_step(float x) {
    int p = __builtin_amdgcn_update_dpp(0, __float_as_int(x), CTRL, 0xf, 0xf, true);
    return fmaxf(x, __int_as_float(p));
}
__device__ __forceinline__ float redmax16(float x) {
    x = dpp_max_step<0x161>(x);
    x = dpp_max_step<0x162>(x);
    x = dpp_max_step<0x164>(x);
    x = dpp_max_step<0x168>(x);
    return x;
}

// A-fragment (16-bit A 16x32) K index for VGPR-pair p, lane-half hi (contiguous: k = 16*hi + 2*p).
__device__ __forceinline__ int a_kbase(int p, int hi) {
    return hi * 16 + 2 * p;
}

__device__ __forceinline__ v16h ones_frag() {
    v16h o;
    #pragma unroll
    for (int e = 0; e < 16; ++e) o[e] = (_Float16)1.0f;
    return o;
}

// ---------------- async global->LDS helpers (CDNA5 ASYNCcnt path) ----------------

__device__ __forceinline__ void async_b128(const void* gsrc, void* ldsdst) {
#if __has_builtin(__builtin_amdgcn_global_load_async_to_lds_b128)
    __builtin_amdgcn_global_load_async_to_lds_b128(
        (__attribute__((address_space(1))) v4i*)(uintptr_t)gsrc,
        (__attribute__((address_space(3))) v4i*)ldsdst, 0, 0);
#else
    unsigned d = (unsigned)(uintptr_t)(__attribute__((address_space(3))) void*)ldsdst;
    unsigned long long g = (unsigned long long)(uintptr_t)gsrc;
    asm volatile("global_load_async_to_lds_b128 %0, %1, off"
                 :: "v"(d), "v"(g) : "memory");
#endif
}

__device__ __forceinline__ void wait_async0() {
#if __has_builtin(__builtin_amdgcn_s_wait_asynccnt)
    __builtin_amdgcn_s_wait_asynccnt(0);
#else
    asm volatile("s_wait_asynccnt 0x0" ::: "memory");
#endif
}

// copy one contiguous 8KB tile (4096 f16) with 128 threads, 4 x b128 each
__device__ __forceinline__ void async_tile_8k(const _Float16* src, void* ldsdst, int tid) {
    const char* s = (const char*)src + tid * 16;
    char* d = (char*)ldsdst + tid * 16;
    #pragma unroll
    for (int i = 0; i < 4; ++i)
        async_b128(s + i * 2048, d + i * 2048);
}

// ---------------- pre-pass 1: K -> f16 tile-row-major, V -> f16 tile-transposed ----------------

__global__ __launch_bounds__(128)
void convert_kv_kernel(const float* __restrict__ K, const float* __restrict__ V,
                       _Float16* __restrict__ Kh, _Float16* __restrict__ Vth, int seq) {
    const int tid = threadIdx.x;
    const int nkt = seq / 32;
    const int b   = blockIdx.x / nkt;
    const int kt  = blockIdx.x % nkt;
    const size_t gbase = ((size_t)b * seq + kt * 32) * HEAD_D;
    const size_t tbase = ((size_t)b * nkt + kt) * (32 * HEAD_D);
    #pragma unroll
    for (int i = 0; i < 8; ++i) {
        const int f   = tid + i * 128;
        const int row = f >> 5;
        const int c4  = f & 31;
        float4 kf = *(const float4*)(K + gbase + (size_t)row * HEAD_D + c4 * 4);
        float4 vf = *(const float4*)(V + gbase + (size_t)row * HEAD_D + c4 * 4);
        h4 kq = { (_Float16)kf.x, (_Float16)kf.y, (_Float16)kf.z, (_Float16)kf.w };
        *(h4*)&Kh[tbase + (size_t)row * HEAD_D + c4 * 4] = kq;
        Vth[tbase + (size_t)(c4 * 4 + 0) * 32 + row] = (_Float16)vf.x;
        Vth[tbase + (size_t)(c4 * 4 + 1) * 32 + row] = (_Float16)vf.y;
        Vth[tbase + (size_t)(c4 * 4 + 2) * 32 + row] = (_Float16)vf.z;
        Vth[tbase + (size_t)(c4 * 4 + 3) * 32 + row] = (_Float16)vf.w;
    }
}

// ---------------- pre-pass 2: per-batch column mean of V (fully-masked-row output) ----------------

__global__ __launch_bounds__(128)
void mean_v_kernel(const float* __restrict__ V, float* __restrict__ meanV, int seq) {
    const int b = blockIdx.x;
    const int d = threadIdx.x;
    const float* Vb = V + (size_t)b * seq * HEAD_D + d;
    float s = 0.f;
    for (int i = 0; i < seq; i += 4) {
        s += Vb[(size_t)(i + 0) * HEAD_D];
        s += Vb[(size_t)(i + 1) * HEAD_D];
        s += Vb[(size_t)(i + 2) * HEAD_D];
        s += Vb[(size_t)(i + 3) * HEAD_D];
    }
    meanV[b * HEAD_D + d] = s * (1.0f / (float)seq);
}

// ---------------- main kernel (async double-buffered fast path) ----------------

__global__ __launch_bounds__(128)
void fa_async_kernel(const float* __restrict__ Q,
                     const _Float16* __restrict__ Kh,
                     const _Float16* __restrict__ Vth,
                     const int* __restrict__ lens,
                     const float* __restrict__ meanV,
                     float* __restrict__ O, int seq) {
    const int tid  = threadIdx.x;
    const int lane = tid & 31;
    const int w    = tid >> 5;
    const int hi   = lane >> 4;
    const int ln   = lane & 15;

    const int blocksPerBatch = (seq / 16) / 4;
    const int b          = blockIdx.x / blocksPerBatch;
    const int blockQBase = (blockIdx.x % blocksPerBatch) * 64;
    const int qBase      = blockQBase + w * 16;
    const int len        = lens[b];
    const int nkt        = seq / 32;
    const int nktLen     = (len + 31) >> 5;
    const int nktEff     = nktLen < nkt ? nktLen : nkt;   // key tiles actually needed
    const bool hasTail   = (len & 31) != 0 && nktEff == nktLen;

    const float* Qb = Q + (size_t)b * seq * HEAD_D;
    float*       Ob = O + (size_t)b * seq * HEAD_D;
    const _Float16* Ktiles = Kh  + (size_t)b * nkt * (32 * HEAD_D);
    const _Float16* Vtiles = Vth + (size_t)b * nkt * (32 * HEAD_D);

    __shared__ _Float16 ldsK[2][32][HEAD_D];   // 2 x 8KB, [key][dim]
    __shared__ _Float16 ldsVt[2][HEAD_D][32];  // 2 x 8KB, [dim][key]
    __shared__ _Float16 ldsP[4][16][32];       // per-wave P patch

    // acc[0..7] = output columns; acc[8] = running row-sum (P * ones)
    v8f acc[9];
    #pragma unroll
    for (int c = 0; c < 9; ++c) acc[c] = (v8f){};
    float mrow[8];
    #pragma unroll
    for (int r = 0; r < 8; ++r) mrow[r] = -__builtin_inff();

    const v16h bOnes = ones_frag();

    if (blockQBase < len) {   // batch-uniform: safe around block barriers
        // ---- Q tile -> 4 A-fragments, scale folded in ----
        const float scale = 0.08838834764831845f; // 1/sqrt(128)
        v16h aQ[4];
        {
            const float* qrow = Qb + (size_t)(qBase + ln) * HEAD_D;
            #pragma unroll
            for (int s = 0; s < 4; ++s) {
                #pragma unroll
                for (int p = 0; p < 8; ++p) {
                    const int kb = a_kbase(p, hi);
                    float2 f = *(const float2*)(qrow + s * 32 + kb);
                    aQ[s][2 * p]     = (_Float16)(f.x * scale);
                    aQ[s][2 * p + 1] = (_Float16)(f.y * scale);
                }
            }
        }

        // tile body; `masked` is a compile-time constant at each inlined call site
        auto tile_body = [&](int kt, int buf, bool masked) {
            const int keyBase = kt * 32;
            v8f s0 = (v8f){}, s1 = (v8f){};
            #pragma unroll
            for (int s = 0; s < 4; ++s) {
                v16h b0, b1;
                #pragma unroll
                for (int p = 0; p < 8; ++p) {
                    const int kr = 16 * hi + 2 * p;
                    h2 x0 = *(const h2*)&ldsK[buf][ln][s * 32 + kr];
                    h2 x1 = *(const h2*)&ldsK[buf][16 + ln][s * 32 + kr];
                    b0[2 * p] = x0[0]; b0[2 * p + 1] = x0[1];
                    b1[2 * p] = x1[0]; b1[2 * p + 1] = x1[1];
                }
                s0 = __builtin_amdgcn_wmma_f32_16x16x32_f16(false, aQ[s], false, b0,
                                                            (short)0, s0, false, false);
                s1 = __builtin_amdgcn_wmma_f32_16x16x32_f16(false, aQ[s], false, b1,
                                                            (short)0, s1, false, false);
            }

            const bool col0 = !masked || (keyBase + ln)      < len;
            const bool col1 = !masked || (keyBase + 16 + ln) < len;
            #pragma unroll
            for (int r = 0; r < 8; ++r) {
                float x0 = s0[r], x1 = s1[r];
                if (masked) {
                    x0 = col0 ? x0 : -1e9f;
                    x1 = col1 ? x1 : -1e9f;
                }
                float tm = redmax16(fmaxf(x0, x1));
                float mnew  = fmaxf(mrow[r], tm);
                float alpha = __expf(mrow[r] - mnew);
                mrow[r] = mnew;
                float p0 = __expf(x0 - mnew);
                float p1 = __expf(x1 - mnew);
                #pragma unroll
                for (int c = 0; c < 9; ++c) acc[c][r] *= alpha;
                ldsP[w][r + 8 * hi][ln]      = (_Float16)p0;
                ldsP[w][r + 8 * hi][16 + ln] = (_Float16)p1;
            }

            v16h aP;
            #pragma unroll
            for (int p = 0; p < 8; ++p) {
                const int kb = a_kbase(p, hi);
                h2 x = *(const h2*)&ldsP[w][ln][kb];
                aP[2 * p] = x[0]; aP[2 * p + 1] = x[1];
            }
            #pragma unroll
            for (int c = 0; c < 8; ++c) {
                v16h bV;
                #pragma unroll
                for (int p = 0; p < 8; ++p) {
                    const int kr = 16 * hi + 2 * p;
                    h2 x = *(const h2*)&ldsVt[buf][c * 16 + ln][kr];
                    bV[2 * p] = x[0]; bV[2 * p + 1] = x[1];
                }
                acc[c] = __builtin_amdgcn_wmma_f32_16x16x32_f16(false, aP, false, bV,
                                                                (short)0, acc[c], false, false);
            }
            acc[8] = __builtin_amdgcn_wmma_f32_16x16x32_f16(false, aP, false, bOnes,
                                                            (short)0, acc[8], false, false);
        };

        // prime buffer 0
        async_tile_8k(Ktiles, &ldsK[0][0][0], tid);
        async_tile_8k(Vtiles, &ldsVt[0][0][0], tid);

        for (int kt = 0; kt < nktEff; ++kt) {
            const int buf = kt & 1;
            wait_async0();        // our copies into ldsK/ldsVt[buf] landed
            __syncthreads();      // everyone's copies landed; prev compute on buf^1 done
            if (kt + 1 < nktEff) {
                async_tile_8k(Ktiles + (size_t)(kt + 1) * (32 * HEAD_D), &ldsK[buf ^ 1][0][0], tid);
                async_tile_8k(Vtiles + (size_t)(kt + 1) * (32 * HEAD_D), &ldsVt[buf ^ 1][0][0], tid);
            }
            if (hasTail && kt == nktEff - 1) tile_body(kt, buf, true);
            else                             tile_body(kt, buf, false);
        }
    }

    // ---- epilogue: valid rows normalized; masked rows get mean(V) exactly ----
    float mv[8];
    #pragma unroll
    for (int c = 0; c < 8; ++c) mv[c] = meanV[b * HEAD_D + c * 16 + ln];
    #pragma unroll
    for (int r = 0; r < 8; ++r) {
        const int row = qBase + r + 8 * hi;
        float* orow = Ob + (size_t)row * HEAD_D;
        if (row < len) {
            const float inv = 1.0f / acc[8][r];
            #pragma unroll
            for (int c = 0; c < 8; ++c)
                orow[c * 16 + ln] = acc[c][r] * inv;
        } else {
            #pragma unroll
            for (int c = 0; c < 8; ++c)
                orow[c * 16 + ln] = mv[c];
        }
    }
}

// ---------------- fallback kernel (no workspace needed) ----------------

__global__ __launch_bounds__(128)
void fa_sync_kernel(const float* __restrict__ Q, const float* __restrict__ K,
                    const float* __restrict__ V, const int* __restrict__ lens,
                    float* __restrict__ O, int seq) {
    const int tid  = threadIdx.x;
    const int lane = tid & 31;
    const int w    = tid >> 5;
    const int hi   = lane >> 4;
    const int ln   = lane & 15;

    const int blocksPerBatch = (seq / 16) / 4;
    const int b     = blockIdx.x / blocksPerBatch;
    const int qBase = ((blockIdx.x % blocksPerBatch) * 4 + w) * 16;
    const int len   = lens[b];

    const float* Qb = Q + (size_t)b * seq * HEAD_D;
    const float* Kb = K + (size_t)b * seq * HEAD_D;
    const float* Vb = V + (size_t)b * seq * HEAD_D;
    float*       Ob = O + (size_t)b * seq * HEAD_D;

    __shared__ _Float16 ldsK[32][HEAD_D];
    __shared__ _Float16 ldsVt[HEAD_D][32];
    __shared__ _Float16 ldsP[4][16][32];

    const float scale = 0.08838834764831845f;
    v16h aQ[4];
    {
        const float* qrow = Qb + (size_t)(qBase + ln) * HEAD_D;
        #pragma unroll
        for (int s = 0; s < 4; ++s)
            #pragma unroll
            for (int p = 0; p < 8; ++p) {
                const int kb = a_kbase(p, hi);
                float2 f = *(const float2*)(qrow + s * 32 + kb);
                aQ[s][2 * p]     = (_Float16)(f.x * scale);
                aQ[s][2 * p + 1] = (_Float16)(f.y * scale);
            }
    }

    const v16h bOnes = ones_frag();

    v8f acc[9];
    #pragma unroll
    for (int c = 0; c < 9; ++c) acc[c] = (v8f){};
    float mrow[8];
    #pragma unroll
    for (int r = 0; r < 8; ++r) mrow[r] = -__builtin_inff();

    const int nkt = seq / 32;
    for (int kt = 0; kt < nkt; ++kt) {
        const int keyBase = kt * 32;
        #pragma unroll
        for (int i = 0; i < 8; ++i) {
            const int f   = tid + i * 128;
            const int row = f >> 5;
            const int c4  = f & 31;
            const size_t goff = (size_t)(keyBase + row) * HEAD_D + c4 * 4;
            float4 kf = *(const float4*)(Kb + goff);
            float4 vf = *(const float4*)(Vb + goff);
            h2 k01 = { (_Float16)kf.x, (_Float16)kf.y };
            h2 k23 = { (_Float16)kf.z, (_Float16)kf.w };
            *(h2*)&ldsK[row][c4 * 4]     = k01;
            *(h2*)&ldsK[row][c4 * 4 + 2] = k23;
            ldsVt[c4 * 4 + 0][row] = (_Float16)vf.x;
            ldsVt[c4 * 4 + 1][row] = (_Float16)vf.y;
            ldsVt[c4 * 4 + 2][row] = (_Float16)vf.z;
            ldsVt[c4 * 4 + 3][row] = (_Float16)vf.w;
        }
        __syncthreads();

        v8f s0 = (v8f){}, s1 = (v8f){};
        #pragma unroll
        for (int s = 0; s < 4; ++s) {
            v16h b0, b1;
            #pragma unroll
            for (int p = 0; p < 8; ++p) {
                const int kr = 16 * hi + 2 * p;
                h2 x0 = *(const h2*)&ldsK[ln][s * 32 + kr];
                h2 x1 = *(const h2*)&ldsK[16 + ln][s * 32 + kr];
                b0[2 * p] = x0[0]; b0[2 * p + 1] = x0[1];
                b1[2 * p] = x1[0]; b1[2 * p + 1] = x1[1];
            }
            s0 = __builtin_amdgcn_wmma_f32_16x16x32_f16(false, aQ[s], false, b0,
                                                        (short)0, s0, false, false);
            s1 = __builtin_amdgcn_wmma_f32_16x16x32_f16(false, aQ[s], false, b1,
                                                        (short)0, s1, false, false);
        }

        const bool col0 = (keyBase + ln)      < len;
        const bool col1 = (keyBase + 16 + ln) < len;
        #pragma unroll
        for (int r = 0; r < 8; ++r) {
            const bool rowOK = (qBase + r + 8 * hi) < len;
            float x0 = (rowOK && col0) ? s0[r] : -1e9f;
            float x1 = (rowOK && col1) ? s1[r] : -1e9f;
            float tm = redmax16(fmaxf(x0, x1));
            float mnew  = fmaxf(mrow[r], tm);
            float alpha = __expf(mrow[r] - mnew);
            mrow[r] = mnew;
            float p0 = __expf(x0 - mnew);
            float p1 = __expf(x1 - mnew);
            #pragma unroll
            for (int c = 0; c < 9; ++c) acc[c][r] *= alpha;
            ldsP[w][r + 8 * hi][ln]      = (_Float16)p0;
            ldsP[w][r + 8 * hi][16 + ln] = (_Float16)p1;
        }

        v16h aP;
        #pragma unroll
        for (int p = 0; p < 8; ++p) {
            const int kb = a_kbase(p, hi);
            h2 x = *(const h2*)&ldsP[w][ln][kb];
            aP[2 * p] = x[0]; aP[2 * p + 1] = x[1];
        }
        #pragma unroll
        for (int c = 0; c < 8; ++c) {
            v16h bV;
            #pragma unroll
            for (int p = 0; p < 8; ++p) {
                const int kr = 16 * hi + 2 * p;
                h2 x = *(const h2*)&ldsVt[c * 16 + ln][kr];
                bV[2 * p] = x[0]; bV[2 * p + 1] = x[1];
            }
            acc[c] = __builtin_amdgcn_wmma_f32_16x16x32_f16(false, aP, false, bV,
                                                            (short)0, acc[c], false, false);
        }
        acc[8] = __builtin_amdgcn_wmma_f32_16x16x32_f16(false, aP, false, bOnes,
                                                        (short)0, acc[8], false, false);
        __syncthreads();
    }

    #pragma unroll
    for (int r = 0; r < 8; ++r) {
        const float inv = 1.0f / acc[8][r];
        float* orow = Ob + (size_t)(qBase + r + 8 * hi) * HEAD_D;
        #pragma unroll
        for (int c = 0; c < 8; ++c)
            orow[c * 16 + ln] = acc[c][r] * inv;
    }
}

extern "C" void kernel_launch(void* const* d_in, const int* in_sizes, int n_in,
                              void* d_out, int out_size, void* d_ws, size_t ws_size,
                              hipStream_t stream) {
    (void)n_in; (void)out_size;
    const float* q    = (const float*)d_in[0];
    const float* k    = (const float*)d_in[1];
    const float* v    = (const float*)d_in[2];
    const int*   lens = (const int*)d_in[3];
    float*       out  = (float*)d_out;

    const int B   = in_sizes[3];
    const int seq = in_sizes[0] / (B * HEAD_D);

    const size_t halfElems = (size_t)B * seq * HEAD_D;
    const size_t meanBytes = (size_t)B * HEAD_D * sizeof(float);
    const size_t need      = halfElems * sizeof(_Float16) * 2 + meanBytes;

    if (ws_size >= need) {
        _Float16* Kh    = (_Float16*)d_ws;
        _Float16* Vth   = Kh + halfElems;
        float*    meanV = (float*)(Vth + halfElems);
        convert_kv_kernel<<<dim3(B * (seq / 32)), 128, 0, stream>>>(k, v, Kh, Vth, seq);
        mean_v_kernel<<<dim3(B), 128, 0, stream>>>(v, meanV, seq);
        fa_async_kernel<<<dim3(B * (seq / 64)), 128, 0, stream>>>(q, Kh, Vth, lens, meanV, out, seq);
    } else {
        fa_sync_kernel<<<dim3(B * (seq / 64)), 128, 0, stream>>>(q, k, v, lens, out, seq);
    }
}
